// BigBirdPegasusSelfAttention_64544768524418
// MI455X (gfx1250) — matrix operand order repaired and checked by reading, not compiled, not run on hardware
//
#include <hip/hip_runtime.h>
#include <hip/hip_bf16.h>
#include <stdint.h>

// ---------------------------------------------------------------------------
// BigBirdPegasus self-attention forward for MI455X (gfx1250, wave32, WMMA).
//   d_in: hidden_states f32[2,2048,1024], mask f32[2,1,1,2048],
//         Wq,bq,Wk,bk,Wv,bv (f32)
//   d_out: ctx f32[2,2048,1024] ++ probs f32[2,16,2048,2048]
// Pipeline:
//   (1) cvt_kernel x4      : X, Wq, Wk, Wv  f32 -> bf16 (one-time, ~40 MB)
//   (2) qkv_kernel         : pure-bf16 WMMA GEMM -> Q,K [B,H,S,Dh], V^T [B,H,Dh,S]
//   (3) attn_kernel        : fused scores (WMMA) + softmax (wave32 shuffles)
//                            + ctx GEMM (WMMA from LDS probs) -- probs only
//                            written once, never re-read from HBM.
// All WMMA loops are software-pipelined (double-buffered fragments) so loads
// for step k+1 issue before the WMMAs of step k.
// ---------------------------------------------------------------------------

#define BNUM 2
#define SDIM 2048
#define DDIM 1024
#define HNUM 16
#define DH   64

typedef __attribute__((ext_vector_type(16))) __bf16 v16bf;
typedef __attribute__((ext_vector_type(8)))  float  v8f;

union FragBF {
    v16bf v;
    unsigned short u[16];
    uint4 q[2];
};

__device__ __forceinline__ unsigned short f2bf(float f) {
    union { __bf16 b; unsigned short u; } c;
    c.b = (__bf16)f;                       // native v_cvt bf16
    return c.u;
}

// A-fragment (MxK = 16x32): lanes 0-15 hold M=lane, K in {klo..klo+7} U {klo+16..klo+23},
// klo = 8*(lane>=16).  bf16 source, row-major.
__device__ __forceinline__ v16bf load_a_bf16(const unsigned short* __restrict__ row, int kk, int lane) {
    const int klo = (lane >> 4) << 3;
    FragBF f;
    f.q[0] = *(const uint4*)(row + kk + klo);        // 8 halves
    f.q[1] = *(const uint4*)(row + kk + klo + 16);   // 8 halves
    return f.v;
}

// B-fragment (KxN = 32x16): lane holds N=lane%16, K in {klo..klo+7} U {klo+16..klo+23}, klo = 8*(lane>=16).
__device__ __forceinline__ v16bf load_b_bf16(const unsigned short* __restrict__ row, int kk, int lane) {
    const int kh = (lane >> 4) << 3;
    FragBF f;
    f.q[0] = *(const uint4*)(row + kk + kh);
    f.q[1] = *(const uint4*)(row + kk + kh + 16);
    return f.v;
}

// A-fragment built from f32 data in LDS (probs strip), converted with v_cvt_pk_bf16_f32.
__device__ __forceinline__ v16bf load_a_lds_f32(const float* srow, int kk, int lane) {
    const int klo = (lane >> 4) << 3;
    const float4 a0 = *(const float4*)(srow + kk + klo);
    const float4 a1 = *(const float4*)(srow + kk + klo + 4);
    const float4 a2 = *(const float4*)(srow + kk + klo + 16);
    const float4 a3 = *(const float4*)(srow + kk + klo + 20);
    const float t[16] = { a0.x,a0.y,a0.z,a0.w, a1.x,a1.y,a1.z,a1.w,
                          a2.x,a2.y,a2.z,a2.w, a3.x,a3.y,a3.z,a3.w };
    v16bf v;
#pragma unroll
    for (int i = 0; i < 16; ++i) v[i] = (__bf16)t[i];
    return v;
}

__device__ __forceinline__ v8f wmma_bf16(v16bf a, v16bf b, v8f c) {
    return __builtin_amdgcn_wmma_f32_16x16x32_bf16(false, a, false, b, (short)0, c, false, false);
}

// ---------------------------------------------------------------------------
// Kernel 0: f32 -> bf16 elementwise convert (8 elements / thread).
// ---------------------------------------------------------------------------
__global__ __launch_bounds__(256)
void cvt_kernel(const float* __restrict__ in, unsigned short* __restrict__ out, int n)
{
    const int i = (blockIdx.x * 256 + threadIdx.x) * 8;
    if (i >= n) return;
    const float4 a = *(const float4*)(in + i);
    const float4 b = *(const float4*)(in + i + 4);
    union { unsigned short u[8]; uint4 q; } o;
    o.u[0] = f2bf(a.x); o.u[1] = f2bf(a.y); o.u[2] = f2bf(a.z); o.u[3] = f2bf(a.w);
    o.u[4] = f2bf(b.x); o.u[5] = f2bf(b.y); o.u[6] = f2bf(b.z); o.u[7] = f2bf(b.w);
    *(uint4*)(out + i) = o.q;
}

// ---------------------------------------------------------------------------
// Kernel 1: QKV projection (pure bf16).  out[m,n] = sum_k X[m,k]*W[n,k] + bias[n]
// Each wave owns a 16(M) x 64(N) strip: 1 A-fragment feeds 4 WMMAs per K-step.
// Fragments double-buffered: next K-step's 5 loads issue before this step's WMMAs.
// z=0 -> Q [B,H,S,Dh] ; z=1 -> K [B,H,S,Dh] ; z=2 -> V^T [B,H,Dh,S]   (all bf16)
// grid (16, 64, 3), block 128 (4 waves; wave = M sub-tile)
// ---------------------------------------------------------------------------
__global__ __launch_bounds__(128)
void qkv_kernel(const unsigned short* __restrict__ Xb,
                const unsigned short* __restrict__ Wqb,
                const unsigned short* __restrict__ Wkb,
                const unsigned short* __restrict__ Wvb,
                const float* __restrict__ bq,
                const float* __restrict__ bk,
                const float* __restrict__ bv,
                unsigned short* __restrict__ qh,
                unsigned short* __restrict__ kh,
                unsigned short* __restrict__ vt)
{
    const int lane = threadIdx.x & 31;
    const int wave = threadIdx.x >> 5;
    const int mt   = blockIdx.y * 4 + wave;   // 0..255
    const int ng   = blockIdx.x;              // 0..15 -> 64 N columns (== one head)
    const int z    = blockIdx.z;              // 0=q 1=k 2=v
    const int m0 = mt * 16, nb = ng * 64;

    const unsigned short* W = (z == 0) ? Wqb : (z == 1) ? Wkb : Wvb;
    const float* bias       = (z == 0) ? bq  : (z == 1) ? bk  : bv;

    const unsigned short* xrow = Xb + (size_t)(m0 + (lane & 15)) * DDIM;
    const unsigned short* wrow[4];
#pragma unroll
    for (int nt = 0; nt < 4; ++nt)
        wrow[nt] = W + (size_t)(nb + nt * 16 + (lane & 15)) * DDIM;

    v8f acc[4] = {};
    v16bf a  = load_a_bf16(xrow,    0, lane);
    v16bf b0 = load_b_bf16(wrow[0], 0, lane);
    v16bf b1 = load_b_bf16(wrow[1], 0, lane);
    v16bf b2 = load_b_bf16(wrow[2], 0, lane);
    v16bf b3 = load_b_bf16(wrow[3], 0, lane);
    for (int kk = 0; kk < DDIM; kk += 32) {
        const int kn = (kk + 32) & (DDIM - 1);      // wraps to 0 on last iter
        const v16bf an  = load_a_bf16(xrow,    kn, lane);
        const v16bf bn0 = load_b_bf16(wrow[0], kn, lane);
        const v16bf bn1 = load_b_bf16(wrow[1], kn, lane);
        const v16bf bn2 = load_b_bf16(wrow[2], kn, lane);
        const v16bf bn3 = load_b_bf16(wrow[3], kn, lane);
        acc[0] = wmma_bf16(a, b0, acc[0]);
        acc[1] = wmma_bf16(a, b1, acc[1]);
        acc[2] = wmma_bf16(a, b2, acc[2]);
        acc[3] = wmma_bf16(a, b3, acc[3]);
        a = an; b0 = bn0; b1 = bn1; b2 = bn2; b3 = bn3;
    }

    const int rsub = (lane >> 4) << 3;
    const int h = ng;                          // nb == h*64
#pragma unroll
    for (int nt = 0; nt < 4; ++nt) {
        const int col = nb + nt * 16 + (lane & 15);
        const int d = col & 63;
        const float bn = bias[col];
#pragma unroll
        for (int r = 0; r < 8; ++r) {
            const int m = m0 + r + rsub;
            const int b_ = m >> 11, s = m & (SDIM - 1);
            const unsigned short hv = f2bf(acc[nt][r] + bn);
            if (z < 2) {
                unsigned short* dst = (z == 0) ? qh : kh;
                dst[(((size_t)b_ * HNUM + h) * SDIM + s) * DH + d] = hv;
            } else {
                vt[(((size_t)b_ * HNUM + h) * DH + d) * SDIM + s] = hv;
            }
        }
    }
}

// ---------------------------------------------------------------------------
// Kernel 2: fused attention for one (b,h) x 16-query strip.
//   Phase A: scores = Q K^T / 8 + mask  (WMMA, 8 waves x 16 key-tiles,
//            K-fragments double-buffered across key tiles)
//            -> 16x2048 f32 strip in LDS (128 KB)
//   Phase B: row softmax (wave32 shuffle reductions); normalized probs
//            written to d_out AND back into LDS
//   Phase C: ctx = probs @ V^T.  Wave w covers K-range [w*256, w*256+256):
//            per step 1 A-fragment (LDS f32 -> bf16) feeds 4 WMMAs,
//            double-buffered.
//   Phase D: 8-way cross-wave reduction of the 16x64 partial tiles through
//            a 32 KB LDS scratch; waves 0..3 store ctx.
// LDS total 160 KB -> 2 blocks per 320 KB WGP.
// grid (128, 32), block 256.
// ---------------------------------------------------------------------------
__global__ __launch_bounds__(256)
void attn_kernel(const unsigned short* __restrict__ qh,
                 const unsigned short* __restrict__ kh,
                 const unsigned short* __restrict__ vt,
                 const float* __restrict__ mask,
                 float* __restrict__ probs,
                 float* __restrict__ ctx)
{
    extern __shared__ float sm[];             // [16][2048] f32 + red [8][4][256]
    float* red = sm + 16 * SDIM;
    const int lane = threadIdx.x & 31;
    const int wave = threadIdx.x >> 5;
    const int mt = blockIdx.x;                // query tile 0..127
    const int bh = blockIdx.y;                // 0..31
    const int b  = bh >> 4, h = bh & 15;
    const int m0 = mt * 16;
    const int rsub = (lane >> 4) << 3;

    // ---------------- Phase A: scores ----------------
    {
        const unsigned short* qrow = qh + (size_t)bh * SDIM * DH
                                        + (size_t)(m0 + (lane & 15)) * DH;
        const unsigned short* kbase = kh + (size_t)bh * SDIM * DH
                                         + (size_t)(lane & 15) * DH;
        const v16bf aq0 = load_a_bf16(qrow, 0, lane);
        const v16bf aq1 = load_a_bf16(qrow, 32, lane);
        const int kt0 = wave * 16;
        v16bf kf0 = load_b_bf16(kbase + (size_t)(kt0 * 16) * DH, 0, lane);
        v16bf kf1 = load_b_bf16(kbase + (size_t)(kt0 * 16) * DH, 32, lane);
        for (int i = 0; i < 16; ++i) {
            const int kt  = kt0 + i;
            const int ktn = kt0 + ((i + 1) & 15);       // wraps on last iter
            const unsigned short* krown = kbase + (size_t)(ktn * 16) * DH;
            const v16bf kn0 = load_b_bf16(krown, 0, lane);
            const v16bf kn1 = load_b_bf16(krown, 32, lane);
            v8f acc = {};
            acc = wmma_bf16(aq0, kf0, acc);
            acc = wmma_bf16(aq1, kf1, acc);
            const int col = kt * 16 + (lane & 15);
            const float mval = mask[b * SDIM + col];
#pragma unroll
            for (int r = 0; r < 8; ++r)
                sm[(r + rsub) * SDIM + col] = acc[r] * 0.125f + mval;
            kf0 = kn0; kf1 = kn1;
        }
    }
    __syncthreads();

    // ---------------- Phase B: softmax ----------------
    {
        float* out = probs + ((size_t)bh * SDIM + m0) * SDIM;
        for (int rr = 0; rr < 2; ++rr) {
            const int row = wave * 2 + rr;
            float* srow = sm + row * SDIM;
            float v[64];
            float mx = -3.402823466e38f;
#pragma unroll
            for (int i = 0; i < 64; ++i) { v[i] = srow[lane + i * 32]; mx = fmaxf(mx, v[i]); }
#pragma unroll
            for (int off = 16; off > 0; off >>= 1) mx = fmaxf(mx, __shfl_xor(mx, off, 32));
            float s = 0.f;
#pragma unroll
            for (int i = 0; i < 64; ++i) { v[i] = __expf(v[i] - mx); s += v[i]; }
#pragma unroll
            for (int off = 16; off > 0; off >>= 1) s += __shfl_xor(s, off, 32);
            const float inv = 1.0f / s;
            float* orow = out + (size_t)row * SDIM;
#pragma unroll
            for (int i = 0; i < 64; ++i) {
                const float p = v[i] * inv;
                orow[lane + i * 32] = p;      // mandatory f32 probs output
                srow[lane + i * 32] = p;      // keep for Phase C
            }
        }
    }
    __syncthreads();

    // ---------------- Phase C: ctx partial GEMM ----------------
    v8f acc[4] = {};
    {
        const unsigned short* vbase = vt + (size_t)bh * DH * SDIM;
        const unsigned short* brow[4];
#pragma unroll
        for (int nt = 0; nt < 4; ++nt)
            brow[nt] = vbase + (size_t)(nt * 16 + (lane & 15)) * SDIM;
        const float* srow = sm + (size_t)(lane & 15) * SDIM;
        const int k0 = wave * 256;
        v16bf a  = load_a_lds_f32(srow, k0, lane);
        v16bf b0 = load_b_bf16(brow[0], k0, lane);
        v16bf b1 = load_b_bf16(brow[1], k0, lane);
        v16bf b2 = load_b_bf16(brow[2], k0, lane);
        v16bf b3 = load_b_bf16(brow[3], k0, lane);
        for (int kk = k0; kk < k0 + 256; kk += 32) {
            const int kn = k0 + ((kk + 32 - k0) & 255);   // wraps on last iter
            const v16bf an  = load_a_lds_f32(srow, kn, lane);
            const v16bf bn0 = load_b_bf16(brow[0], kn, lane);
            const v16bf bn1 = load_b_bf16(brow[1], kn, lane);
            const v16bf bn2 = load_b_bf16(brow[2], kn, lane);
            const v16bf bn3 = load_b_bf16(brow[3], kn, lane);
            acc[0] = wmma_bf16(a, b0, acc[0]);
            acc[1] = wmma_bf16(a, b1, acc[1]);
            acc[2] = wmma_bf16(a, b2, acc[2]);
            acc[3] = wmma_bf16(a, b3, acc[3]);
            a = an; b0 = bn0; b1 = bn1; b2 = bn2; b3 = bn3;
        }
    }
    // write partials to LDS scratch
#pragma unroll
    for (int nt = 0; nt < 4; ++nt)
#pragma unroll
        for (int r = 0; r < 8; ++r)
            red[((wave * 4 + nt) * 256) + r * 32 + lane] = acc[nt][r];
    __syncthreads();

    // ---------------- Phase D: reduce + store ----------------
    if (wave < 4) {
        const int nt = wave;
        const int d = nt * 16 + (lane & 15);
#pragma unroll
        for (int r = 0; r < 8; ++r) {
            float s = 0.f;
#pragma unroll
            for (int w8 = 0; w8 < 8; ++w8)
                s += red[((w8 * 4 + nt) * 256) + r * 32 + lane];
            const int m = m0 + r + rsub;       // query row s
            ctx[((size_t)(b * SDIM + m)) * DDIM + h * DH + d] = s;
        }
    }
}

// ---------------------------------------------------------------------------
extern "C" void kernel_launch(void* const* d_in, const int* in_sizes, int n_in,
                              void* d_out, int out_size, void* d_ws, size_t ws_size,
                              hipStream_t stream) {
    (void)in_sizes; (void)n_in; (void)out_size; (void)ws_size;
    const float* X    = (const float*)d_in[0];
    const float* mask = (const float*)d_in[1];
    const float* Wq   = (const float*)d_in[2];
    const float* bq   = (const float*)d_in[3];
    const float* Wk   = (const float*)d_in[4];
    const float* bk   = (const float*)d_in[5];
    const float* Wv   = (const float*)d_in[6];
    const float* bv   = (const float*)d_in[7];

    float* out   = (float*)d_out;
    float* ctx   = out;                                     // [2,2048,1024]
    float* probs = out + (size_t)BNUM * SDIM * DDIM;        // [2,16,2048,2048]

    const size_t qkv_elems = (size_t)BNUM * HNUM * SDIM * DH;   // 4,194,304
    const size_t x_elems   = (size_t)BNUM * SDIM * DDIM;        // 4,194,304
    const size_t w_elems   = (size_t)DDIM * DDIM;               // 1,048,576
    unsigned short* qh  = (unsigned short*)d_ws;
    unsigned short* kh  = qh + qkv_elems;
    unsigned short* vt  = kh + qkv_elems;
    unsigned short* xb  = vt + qkv_elems;
    unsigned short* wqb = xb + x_elems;
    unsigned short* wkb = wqb + w_elems;
    unsigned short* wvb = wkb + w_elems;                    // total ws: 38 MB

    cvt_kernel<<<dim3((unsigned)(x_elems / 2048)), 256, 0, stream>>>(X, xb, (int)x_elems);
    cvt_kernel<<<dim3((unsigned)(w_elems / 2048)), 256, 0, stream>>>(Wq, wqb, (int)w_elems);
    cvt_kernel<<<dim3((unsigned)(w_elems / 2048)), 256, 0, stream>>>(Wk, wkb, (int)w_elems);
    cvt_kernel<<<dim3((unsigned)(w_elems / 2048)), 256, 0, stream>>>(Wv, wvb, (int)w_elems);

    qkv_kernel<<<dim3(16, 64, 3), 128, 0, stream>>>(xb, wqb, wkb, wvb, bq, bk, bv, qh, kh, vt);

    const size_t lds_bytes = (size_t)(16 * SDIM + 8 * 4 * 256) * sizeof(float); // 160 KB
    attn_kernel<<<dim3(128, 32), 256, lds_bytes, stream>>>(qh, kh, vt, mask, probs, ctx);
}
